// MResConv_76141180223547
// MI455X (gfx1250) — compile-verified, matrix operand order checked by reading, not run on hardware
//
#include <hip/hip_runtime.h>
#include <hip/hip_bf16.h>

// ---------------------------------------------------------------------------
// MResConv on gfx1250: mesh-conv as bf16 WMMA GEMM with fused gather/BN/resid.
// LDS panels are stored in WMMA-fragment-ready layout so fragment loads are
// two ds_load_b128 per fragment (instead of 8 scattered b32 loads).
// ---------------------------------------------------------------------------

typedef __attribute__((ext_vector_type(16))) __bf16 v16bf;
typedef __attribute__((ext_vector_type(8)))  float  v8f;

#define BATCH   8
#define EDGES   16384
#define OCH     256
#define K7      7
#define M_TILE  64
#define N_TILE  64
#define KC      32        // WMMA K step (channels per chunk)
#define MF      (M_TILE / 16)
#define NF      (N_TILE / 16)
#define NEG_SLOPE 0.01f
#define BN_EPS  1e-5f

union FragBF { v16bf v; uint4 q[2]; unsigned int u[8]; };
union FragF  { v8f   v; float f[8]; };

__device__ __forceinline__ unsigned short f2bf(float f) {
    union { float f; unsigned int u; } c; c.f = f;
    unsigned int u = c.u;
    u += 0x7FFFu + ((u >> 16) & 1u);          // round-to-nearest-even
    return (unsigned short)(u >> 16);
}
__device__ __forceinline__ unsigned int pack2bf(float lo, float hi) {
    return (unsigned int)f2bf(lo) | ((unsigned int)f2bf(hi) << 16);
}
__device__ __forceinline__ float lrelu(float v) {
    return v > 0.f ? v : NEG_SLOPE * v;
}

// Fragment-layout position of channel-offset kc within a 16x32 bf16 panel:
//   half = (kc>>3)&1 ; elem i = (kc&7) + 8*(kc>>4) ; lane slot = 16*half + lm
__device__ __forceinline__ int frag_elem(int kc) {
    return (kc & 7) + 8 * (kc >> 4);
}
__device__ __forceinline__ int frag_half(int kc) {
    return (kc >> 3) & 1;
}

// ---------------------------------------------------------------------------
// Tiled mesh-conv GEMM.  out[b,o,e] = sum_{c,k7} G_k7[b,c,e] * W[o,c,k7]
// ---------------------------------------------------------------------------
template <int CIN, bool BN_IN, bool RES_OUT>
__global__ __launch_bounds__(256)
void meshconv_wmma(const float* __restrict__ X,        // [B,CIN,E]
                   const int*   __restrict__ gemmIdx,  // [B,E,4]
                   const float* __restrict__ W,        // [OCH,CIN,7]
                   const float* __restrict__ scale,    // [OCH] (BN_IN)
                   const float* __restrict__ shift,    // [OCH] (BN_IN)
                   const float* __restrict__ residual, // [B,OCH,E] (RES_OUT)
                   float*       __restrict__ out)      // [B,OCH,E]
{
    // Fragment-ready panels: [k7][frag16][lane(32)][elem(16)] bf16
    __shared__ __align__(16) unsigned short AtileF[K7][MF][32][16];
    __shared__ __align__(16) unsigned short BtileF[K7][NF][32][16];
    __shared__ int nbr[N_TILE][4];

    const int b    = blockIdx.z;
    const int m0   = blockIdx.y * M_TILE;
    const int e0   = blockIdx.x * N_TILE;
    const int tid  = threadIdx.x;
    const int lane = tid & 31;
    const int wave = tid >> 5;
    const int wm   = wave & 3;   // M fragment index (16 rows)
    const int wn   = wave >> 2;  // N half (2 fragments of 16 cols)
    const int half = lane >> 4;
    const int lm   = lane & 15;

    // Stage neighbor indices for this edge tile (one int4 per edge).
    if (tid < N_TILE) {
        const int4 g = *(const int4*)(gemmIdx + ((size_t)b * EDGES + (e0 + tid)) * 4);
        nbr[tid][0] = g.x; nbr[tid][1] = g.y; nbr[tid][2] = g.z; nbr[tid][3] = g.w;
    }

    FragF acc[2];
#pragma unroll
    for (int j = 0; j < 2; ++j)
#pragma unroll
        for (int r = 0; r < 8; ++r) acc[j].f[r] = 0.f;

    for (int c0 = 0; c0 < CIN; c0 += KC) {
        __syncthreads();   // nbr visible (iter 0) / previous frag reads done

        // ---- A panels: each thread handles a channel *pair* -> packed u32 ----
        for (int p = tid; p < M_TILE * (KC / 2); p += 256) {
            const int m   = p >> 4;
            const int kc  = (p & 15) * 2;          // even channel offset
            const int mf  = m >> 4;
            const int slot = 16 * frag_half(kc) + (m & 15);
            const int ie  = frag_elem(kc);          // even elem; pair is (ie, ie+1)
            const float* wp0 = W + ((size_t)(m0 + m) * CIN + (c0 + kc)) * K7;
            const float* wp1 = wp0 + K7;
#pragma unroll
            for (int k = 0; k < K7; ++k)
                *(unsigned int*)&AtileF[k][mf][slot][ie] = pack2bf(wp0[k], wp1[k]);
        }

        // ---- B panels: gather 2x5 values, build 2x7 features, packed u32 ----
        for (int p = tid; p < N_TILE * (KC / 2); p += 256) {
            const int n    = p >> 4;
            const int kc   = (p & 15) * 2;
            const int nf   = n >> 4;
            const int slot = 16 * frag_half(kc) + (n & 15);
            const int ie   = frag_elem(kc);
            const int c    = c0 + kc;
            const int i0 = e0 + n, i1 = nbr[n][0], i2 = nbr[n][1],
                      i3 = nbr[n][2], i4 = nbr[n][3];
            float feat[2][K7];
#pragma unroll
            for (int cc = 0; cc < 2; ++cc) {
                const float* xc = X + ((size_t)b * CIN + (c + cc)) * EDGES;
                float f0 = xc[i0], f1 = xc[i1], f2 = xc[i2], f3 = xc[i3], f4 = xc[i4];
                if (BN_IN) {
                    const float sc = scale[c + cc], sh = shift[c + cc];
                    f0 = lrelu(f0) * sc + sh;
                    f1 = lrelu(f1) * sc + sh;
                    f2 = lrelu(f2) * sc + sh;
                    f3 = lrelu(f3) * sc + sh;
                    f4 = lrelu(f4) * sc + sh;
                }
                const float x5 = f1 + f2 + f3 + f4;
                const float av = 0.25f * x5;
                const float d1 = f1 - av, d2 = f2 - av, d3 = f3 - av, d4 = f4 - av;
                feat[cc][0] = f0;
                feat[cc][1] = f1 + f3;
                feat[cc][2] = f2 + f4;
                feat[cc][3] = fabsf(f1 - f3);
                feat[cc][4] = fabsf(f2 - f4);
                feat[cc][5] = x5;
                feat[cc][6] = d1 * d1 + d2 * d2 + d3 * d3 + d4 * d4;
            }
#pragma unroll
            for (int k = 0; k < K7; ++k)
                *(unsigned int*)&BtileF[k][nf][slot][ie] = pack2bf(feat[0][k], feat[1][k]);
        }
        __syncthreads();

        // ---- WMMA over the 7 feature slices: 2x ds_load_b128 per fragment ----
#pragma unroll
        for (int k7 = 0; k7 < K7; ++k7) {
            FragBF afr, bfr[2];
            afr.q[0] = *(const uint4*)&AtileF[k7][wm][lane][0];
            afr.q[1] = *(const uint4*)&AtileF[k7][wm][lane][8];
#pragma unroll
            for (int fn = 0; fn < 2; ++fn) {
                bfr[fn].q[0] = *(const uint4*)&BtileF[k7][wn * 2 + fn][lane][0];
                bfr[fn].q[1] = *(const uint4*)&BtileF[k7][wn * 2 + fn][lane][8];
            }
#pragma unroll
            for (int fn = 0; fn < 2; ++fn)
                acc[fn].v = __builtin_amdgcn_wmma_f32_16x16x32_bf16(
                    false, afr.v, false, bfr[fn].v,
                    (short)0, acc[fn].v, false, false);
        }
    }

    // ---- Epilogue: f32 C/D layout -> row = r + 8*half, col = lm ----
#pragma unroll
    for (int fn = 0; fn < 2; ++fn) {
        const int col  = e0 + wn * 32 + fn * 16 + lm;
        const int mrow = m0 + wm * 16 + 8 * half;
        const size_t base = ((size_t)b * OCH + mrow) * EDGES + col;
#pragma unroll
        for (int r = 0; r < 8; ++r) {
            const size_t idx = base + (size_t)r * EDGES;
            float v = acc[fn].f[r];
            if (RES_OUT) {
                v += residual[idx];
                v = lrelu(v);
            }
            out[idx] = v;
        }
    }
}

// ---------------------------------------------------------------------------
// BN statistics over leaky_relu(h): one block per channel, folds gamma/beta
// into a fused affine (scale, shift).
// ---------------------------------------------------------------------------
__global__ __launch_bounds__(256)
void bn_stats(const float* __restrict__ h, const float* __restrict__ gamma,
              const float* __restrict__ beta, float* __restrict__ scale,
              float* __restrict__ shift)
{
    const int c = blockIdx.x, tid = threadIdx.x;
    float s = 0.f, s2 = 0.f;
    for (int b = 0; b < BATCH; ++b) {
        const float* p = h + ((size_t)b * OCH + c) * EDGES;
        for (int e = tid; e < EDGES; e += 256) {
            float v = lrelu(p[e]);
            s += v; s2 += v * v;
        }
    }
    __shared__ float rs[256], rq[256];
    rs[tid] = s; rq[tid] = s2;
    __syncthreads();
    for (int off = 128; off > 0; off >>= 1) {
        if (tid < off) { rs[tid] += rs[tid + off]; rq[tid] += rq[tid + off]; }
        __syncthreads();
    }
    if (tid == 0) {
        const float n = (float)BATCH * (float)EDGES;
        float mean = rs[0] / n;
        float var  = rq[0] / n - mean * mean;
        float sc   = gamma[c] * rsqrtf(var + BN_EPS);
        scale[c] = sc;
        shift[c] = beta[c] - mean * sc;
    }
}

// ---------------------------------------------------------------------------
extern "C" void kernel_launch(void* const* d_in, const int* in_sizes, int n_in,
                              void* d_out, int out_size, void* d_ws, size_t ws_size,
                              hipStream_t stream) {
    const float* x     = (const float*)d_in[0];  // [8,128,16384,1]
    const int*   gemmI = (const int*)  d_in[1];  // [8,16384,4]
    const float* W0    = (const float*)d_in[2];  // [256,128,7]
    const float* W1    = (const float*)d_in[3];  // [256,256,7]
    const float* gamma = (const float*)d_in[4];  // [256]
    const float* beta  = (const float*)d_in[5];  // [256]
    float* out = (float*)d_out;                  // [8,256,16384,1]

    float* h     = (float*)d_ws;                           // [8,256,16384] f32
    float* scale = h + (size_t)BATCH * OCH * EDGES;        // [256]
    float* shift = scale + OCH;                            // [256]

    dim3 blk(256);
    dim3 grid(EDGES / N_TILE, OCH / M_TILE, BATCH);

    // conv0: h = meshconv(x, W0)
    meshconv_wmma<128, false, false><<<grid, blk, 0, stream>>>(
        x, gemmI, W0, nullptr, nullptr, nullptr, h);

    // BN stats on leaky_relu(h) -> fused affine
    bn_stats<<<OCH, blk, 0, stream>>>(h, gamma, beta, scale, shift);

    // conv1 with fused (lrelu+BN) on gathered inputs, residual + lrelu epilogue
    meshconv_wmma<256, true, true><<<grid, blk, 0, stream>>>(
        h, gemmI, W1, scale, shift, h, out);
}
